// HebbianGroupMemory_80719615361693
// MI455X (gfx1250) — matrix operand order repaired and checked by reading, not compiled, not run on hardware
//
#include <hip/hip_runtime.h>
#include <hip/hip_bf16.h>

// Problem sizes (fixed by the reference)
#define NTOK 16384      // 8*2048 flattened embeddings
#define DIM  1024
#define NG   4096
#define DECAY 0.97f
#define TEMPR 0.7f

typedef __attribute__((ext_vector_type(16))) __bf16 v16bf;
typedef __attribute__((ext_vector_type(8)))  __bf16 v8bf;
typedef __attribute__((ext_vector_type(8)))  float  v8f;

union BF16x16 { v16bf v; v8bf h[2]; __bf16 e[16]; };

// ---------------- workspace layout (bytes) ----------------
// qn bf16 [N,D]  (reused as flatT bf16 [D,N] after GEMM1): 33554432
// pn bf16 [G,D]:                                            8388608
// homeo f32 [G]:                                              16384
// wsum  f32 [G]:                                              16384
// wT bf16 [G,N]:                                          134217728
#define QN_OFF    0ull
#define PN_OFF    33554432ull
#define HOMEO_OFF 41943040ull
#define WSUM_OFF  41959424ull
#define WT_OFF    41975808ull

#define WMMA_BF16(ACC, A, B) \
    __builtin_amdgcn_wmma_f32_16x16x32_bf16(false, (A).v, false, (B).v, (short)0, (ACC), false, false)

// ---------------------------------------------------------------------------
// Row l2-normalize f32 -> bf16 (one block / row, cols == 1024)
__global__ void normalize_rows_bf16(const float* __restrict__ src,
                                    __bf16* __restrict__ dst, int cols) {
    __shared__ float red[256];
    const float* row = src + (size_t)blockIdx.x * cols;
    __bf16* orow = dst + (size_t)blockIdx.x * cols;
    const int t = threadIdx.x;
    float s = 0.0f;
    for (int i = t * 4; i < cols; i += 1024) {
        float4 v = *(const float4*)(row + i);
        s += v.x * v.x + v.y * v.y + v.z * v.z + v.w * v.w;
    }
    red[t] = s; __syncthreads();
    for (int st = 128; st > 0; st >>= 1) {
        if (t < st) red[t] += red[t + st];
        __syncthreads();
    }
    const float scale = 1.0f / fmaxf(sqrtf(red[0]), 1e-12f);
    for (int i = t * 4; i < cols; i += 1024) {
        float4 v = *(const float4*)(row + i);
        orow[i + 0] = (__bf16)(v.x * scale);
        orow[i + 1] = (__bf16)(v.y * scale);
        orow[i + 2] = (__bf16)(v.z * scale);
        orow[i + 3] = (__bf16)(v.w * scale);
    }
}

__global__ void homeo_kernel(const float* __restrict__ usage,
                             float* __restrict__ homeo) {
    int g = blockIdx.x * blockDim.x + threadIdx.x;
    if (g < NG) homeo[g] = logf(fmaxf(usage[g], 1e-6f));
}

__global__ void zero_kernel(float* __restrict__ p) {
    int i = blockIdx.x * blockDim.x + threadIdx.x;
    if (i < NG) p[i] = 0.0f;
}

// ---------------------------------------------------------------------------
// GEMM1: logits[n,g] = sum_k qn[n,k] * pn[g,k]  (both K-contiguous, NT)
// Block = 8 waves sharing the same 64-wide B strip (B fragments hit L0/L1);
// each wave owns a 16(M) x 64(N) tile. One A pointer + one B base pointer;
// the 4 B rows are reached via compile-time immediate offsets (j*16*DIM
// elements = 32KB steps, inside the 24-bit global-load offset field), which
// keeps address pressure low so the compiler can unroll/pipeline the loop.
// Epilogue fuses (logit - homeo[g]) / TEMP.
__global__ void gemm1_wmma(const __bf16* __restrict__ qn,
                           const __bf16* __restrict__ pn,
                           const float* __restrict__ homeo,
                           float* __restrict__ out) {
    const int lane = threadIdx.x & 31;
    const int wv   = threadIdx.x >> 5;            // 0..7
    const int m0 = blockIdx.x * 128 + wv * 16;
    const int n0 = blockIdx.y * 64;
    const int row  = lane & 15;
    const int half = lane >> 4;

    const __bf16* aRow  = qn + (size_t)(m0 + row) * DIM;
    const __bf16* bBase = pn + (size_t)(n0 + row) * DIM + half * 16;

    v8f acc0 = {}, acc1 = {}, acc2 = {}, acc3 = {};

    for (int k = 0; k < DIM; k += 32) {
        __builtin_prefetch(aRow + k + 64, 0, 3);   // global_prefetch_b8
        BF16x16 a, b0, b1, b2, b3;
        // A-layout (matches B): lanes 0-15 hold K{0..15}, lanes 16-31 K{16..31}
        a.h[0] = *(const v8bf*)(aRow + k + half * 16);
        a.h[1] = *(const v8bf*)(aRow + k + 8 + half * 16);
        b0.v = *(const v16bf*)(bBase + k);
        b1.v = *(const v16bf*)(bBase + 16 * DIM + k);
        b2.v = *(const v16bf*)(bBase + 32 * DIM + k);
        b3.v = *(const v16bf*)(bBase + 48 * DIM + k);
        acc0 = WMMA_BF16(acc0, a, b0);
        acc1 = WMMA_BF16(acc1, a, b1);
        acc2 = WMMA_BF16(acc2, a, b2);
        acc3 = WMMA_BF16(acc3, a, b3);
    }

    const float invT = 1.0f / TEMPR;
    v8f accs[4] = {acc0, acc1, acc2, acc3};
#pragma unroll
    for (int j = 0; j < 4; ++j) {
        const int n = n0 + j * 16 + row;
        const float h = homeo[n];
#pragma unroll
        for (int r = 0; r < 8; ++r) {
            const int m = m0 + r + half * 8;
            out[(size_t)m * NG + n] = (accs[j][r] - h) * invT;
        }
    }
}

// ---------------------------------------------------------------------------
// Row softmax in place (one block / row, G = 4096 elements)
__global__ void softmax_rows(float* __restrict__ w) {
    __shared__ float red[256];
    float* row = w + (size_t)blockIdx.x * NG;
    const int t = threadIdx.x;
    float m = -3.402823466e38f;
    for (int i = t; i < NG; i += 256) m = fmaxf(m, row[i]);
    red[t] = m; __syncthreads();
    for (int s = 128; s > 0; s >>= 1) {
        if (t < s) red[t] = fmaxf(red[t], red[t + s]);
        __syncthreads();
    }
    m = red[0]; __syncthreads();
    float sum = 0.0f;
    for (int i = t; i < NG; i += 256) {
        float e = __expf(row[i] - m);
        row[i] = e;
        sum += e;
    }
    red[t] = sum; __syncthreads();
    for (int s = 128; s > 0; s >>= 1) {
        if (t < s) red[t] += red[t + s];
        __syncthreads();
    }
    const float inv = 1.0f / red[0];
    for (int i = t; i < NG; i += 256) row[i] *= inv;
}

// ---------------------------------------------------------------------------
// Column-sum partials with f32 atomics: wsum[g] = sum_n W[n,g]
__global__ void colsum_partial(const float* __restrict__ w,
                               float* __restrict__ wsum, int rowsPer) {
    const int g = blockIdx.x * blockDim.x + threadIdx.x;
    const int r0 = blockIdx.y * rowsPer;
    float s = 0.0f;
    for (int n = r0; n < r0 + rowsPer; ++n)
        s += w[(size_t)n * NG + g];
    atomicAdd(&wsum[g], s);
}

__global__ void usage_update(const float* __restrict__ usage,
                             const float* __restrict__ wsum,
                             float* __restrict__ uo) {
    int g = blockIdx.x * blockDim.x + threadIdx.x;
    if (g >= NG) return;
    const float s = wsum[g];
    const float u = usage[g] * DECAY;
    uo[g] = (s > 0.0f) ? u + (1.0f - DECAY) * s : u;
}

// ---------------------------------------------------------------------------
// LDS-tiled transpose + convert: src[N, srcCols] f32 -> dst[srcCols, N] bf16.
// 32x32 tiles, block (32,8). Used for W[N,G]->wT[G,N] and pair[N,D]->flatT[D,N].
__global__ void transpose_convert(const float* __restrict__ src,
                                  __bf16* __restrict__ dst,
                                  int srcCols, int srcRows) {
    __shared__ float tile[32][33];
    const int c0 = blockIdx.x * 32;
    const int n0 = blockIdx.y * 32;
    const int tx = threadIdx.x;   // 0..31
    const int ty = threadIdx.y;   // 0..7
#pragma unroll
    for (int r = 0; r < 4; ++r) {
        const int n = n0 + ty + r * 8;
        tile[ty + r * 8][tx] = src[(size_t)n * srcCols + c0 + tx];
    }
    __syncthreads();
#pragma unroll
    for (int r = 0; r < 4; ++r) {
        const int c = c0 + ty + r * 8;
        dst[(size_t)c * srcRows + n0 + tx] = (__bf16)tile[tx][ty + r * 8];
    }
}

// ---------------------------------------------------------------------------
// GEMM2: upd[g,d] = sum_n wT[g,n] * flatT[d,n]  (both K-contiguous after the
// transposes -> identical NT microkernel as GEMM1, K = NTOK). Block = 8 waves
// sharing one 64-wide d-strip; each wave one 16(g) x 64(d) tile. Single B
// base + immediate offsets (j*16*NTOK elements = 512KB steps, still within
// the 24-bit offset field).
__global__ void gemm2_wmma(const __bf16* __restrict__ wT,
                           const __bf16* __restrict__ flatT,
                           float* __restrict__ upd) {
    const int lane = threadIdx.x & 31;
    const int wv   = threadIdx.x >> 5;            // 0..7
    const int g0 = blockIdx.x * 128 + wv * 16;
    const int d0 = blockIdx.y * 64;
    const int row  = lane & 15;
    const int half = lane >> 4;

    const __bf16* aRow  = wT    + (size_t)(g0 + row) * NTOK;
    const __bf16* bBase = flatT + (size_t)(d0 + row) * NTOK + half * 16;

    v8f acc0 = {}, acc1 = {}, acc2 = {}, acc3 = {};

    for (int k = 0; k < NTOK; k += 32) {
        __builtin_prefetch(aRow + k + 64, 0, 3);
        BF16x16 a, b0, b1, b2, b3;
        a.h[0] = *(const v8bf*)(aRow + k + half * 16);
        a.h[1] = *(const v8bf*)(aRow + k + 8 + half * 16);
        b0.v = *(const v16bf*)(bBase + k);
        b1.v = *(const v16bf*)(bBase + 16 * NTOK + k);
        b2.v = *(const v16bf*)(bBase + 32 * NTOK + k);
        b3.v = *(const v16bf*)(bBase + 48 * NTOK + k);
        acc0 = WMMA_BF16(acc0, a, b0);
        acc1 = WMMA_BF16(acc1, a, b1);
        acc2 = WMMA_BF16(acc2, a, b2);
        acc3 = WMMA_BF16(acc3, a, b3);
    }

    v8f accs[4] = {acc0, acc1, acc2, acc3};
#pragma unroll
    for (int j = 0; j < 4; ++j) {
#pragma unroll
        for (int r = 0; r < 8; ++r) {
            const int g = g0 + r + half * 8;
            const int d = d0 + j * 16 + row;
            upd[(size_t)g * DIM + d] = accs[j][r];
        }
    }
}

// ---------------------------------------------------------------------------
// Finalize patterns: upd /= (wsum+eps); blend; l2norm; where(valid)
__global__ void finalize_patterns(const float* __restrict__ pat,
                                  const float* __restrict__ wsum,
                                  float* __restrict__ po) {
    __shared__ float red[256];
    const int g = blockIdx.x;
    const float ws = wsum[g];
    const bool valid = ws > 0.0f;
    const float inv = 1.0f / (ws + 1e-6f);
    const float* prow = pat + (size_t)g * DIM;
    float* orow = po + (size_t)g * DIM;
    const int t = threadIdx.x;
    float b[4];
    float s = 0.0f;
#pragma unroll
    for (int i = 0; i < 4; ++i) {
        const int idx = t + i * 256;
        const float u = orow[idx] * inv;        // raw sums staged by gemm2
        const float bb = prow[idx] * DECAY + (1.0f - DECAY) * u;
        b[i] = bb;
        s += bb * bb;
    }
    red[t] = s; __syncthreads();
    for (int st = 128; st > 0; st >>= 1) {
        if (t < st) red[t] += red[t + st];
        __syncthreads();
    }
    const float scale = 1.0f / fmaxf(sqrtf(red[0]), 1e-12f);
#pragma unroll
    for (int i = 0; i < 4; ++i) {
        const int idx = t + i * 256;
        orow[idx] = valid ? b[i] * scale : prow[idx];
    }
}

// ---------------------------------------------------------------------------
extern "C" void kernel_launch(void* const* d_in, const int* in_sizes, int n_in,
                              void* d_out, int out_size, void* d_ws, size_t ws_size,
                              hipStream_t stream) {
    (void)in_sizes; (void)n_in; (void)out_size; (void)ws_size;
    const float* pair  = (const float*)d_in[0];   // [8,2048,1024]
    const float* pats  = (const float*)d_in[1];   // [4096,1024]
    const float* usage = (const float*)d_in[2];   // [4096]

    float* out  = (float*)d_out;
    float* wout = out;                                   // weights [N,G]
    float* pout = out + (size_t)NTOK * NG;               // patterns_new [G,D]
    float* uout = pout + (size_t)NG * DIM;               // usage_new [G]

    char* ws = (char*)d_ws;
    __bf16* qn    = (__bf16*)(ws + QN_OFF);   // reused as flatT after GEMM1
    __bf16* pn    = (__bf16*)(ws + PN_OFF);
    float*  homeo = (float*)(ws + HOMEO_OFF);
    float*  wsum  = (float*)(ws + WSUM_OFF);
    __bf16* wT    = (__bf16*)(ws + WT_OFF);

    // 1) normalize + convert to bf16
    normalize_rows_bf16<<<NTOK, 256, 0, stream>>>(pair, qn, DIM);
    normalize_rows_bf16<<<NG,   256, 0, stream>>>(pats, pn, DIM);
    homeo_kernel<<<NG / 256, 256, 0, stream>>>(usage, homeo);
    zero_kernel <<<NG / 256, 256, 0, stream>>>(wsum);

    // 2) GEMM1 (WMMA bf16): blocks of 8 waves sharing the B strip
    gemm1_wmma<<<dim3(NTOK / 128, NG / 64), 256, 0, stream>>>(qn, pn, homeo, wout);

    // 3) row softmax in place
    softmax_rows<<<NTOK, 256, 0, stream>>>(wout);

    // 4) column sums + usage EMA
    colsum_partial<<<dim3(NG / 256, 32), 256, 0, stream>>>(wout, wsum, NTOK / 32);
    usage_update<<<NG / 256, 256, 0, stream>>>(usage, wsum, uout);

    // 5) transposes: W[N,G] -> wT[G,N] bf16; pair[N,D] -> flatT[D,N] bf16
    transpose_convert<<<dim3(NG / 32, NTOK / 32), dim3(32, 8), 0, stream>>>(wout, wT, NG, NTOK);
    transpose_convert<<<dim3(DIM / 32, NTOK / 32), dim3(32, 8), 0, stream>>>(pair, qn, DIM, NTOK);

    // 6) GEMM2 (WMMA bf16, NT after transposes): raw updates -> patterns region
    gemm2_wmma<<<dim3(NG / 128, DIM / 64), 256, 0, stream>>>(wT, qn, pout);

    // 7) finalize patterns (EMA blend + l2norm + valid mask)
    finalize_patterns<<<NG, 256, 0, stream>>>(pats, wsum, pout);
}
